// critic_crossattn_27771258536167
// MI455X (gfx1250) — matrix operand order repaired and checked by reading, not compiled, not run on hardware
//
#include <hip/hip_runtime.h>
#include <hip/hip_bf16.h>

typedef __bf16 bf16;
typedef __attribute__((ext_vector_type(16))) __bf16 v16bf;
typedef __attribute__((ext_vector_type(8)))  __bf16 v8bf;
typedef __attribute__((ext_vector_type(8)))  float   v8f;
typedef __attribute__((ext_vector_type(4))) unsigned int u32x4;
typedef __attribute__((ext_vector_type(4))) int i32x4;
typedef __attribute__((ext_vector_type(8))) int i32x8;

#define WAVES 2          // waves per block, each wave owns 16 batch rows
#define ROWS_PER_BLOCK (WAVES * 16)
#define BATCH 65536
#define N_OBJ 6

#if defined(__HIP_DEVICE_COMPILE__) && __has_builtin(__builtin_amdgcn_tensor_load_to_lds)
#define USE_TDM 1
#else
#define USE_TDM 0
#endif

// ---- bf16 weight workspace layout (element offsets) ----
#define WS_RW1(p) ((p)*8192 + 0)      // [64][32]  (K=14 padded to 32)
#define WS_RW2(p) ((p)*8192 + 2048)   // [32][64]
#define WS_OW1(p) ((p)*8192 + 4096)   // [64][32]  (K=18 padded to 32)
#define WS_OW2(p) ((p)*8192 + 6144)   // [32][64]
#define WS_QW1    16384               // [128][128]
#define WS_QW2    32768               // [128][128]

struct Params {
  const float* x; const float* act;
  const float* rb1[2]; const float* rb2[2];
  const float* ob1[2]; const float* ob2[2];
  const float* g[2];   const float* b[2];
  const float* qb1; const float* qb2; const float* qw3; const float* qb3;
  float* out;
};

struct PrepParams {
  const float* rw1[2]; const float* rw2[2];
  const float* ow1[2]; const float* ow2[2];
  const float* qw1; const float* qw2;
};

// ---------------- fragment helpers (CDNA5 WMMA bf16 16x16x32) ----------------

// A-matrix 16x32 bf16: lane m=l&15; lanes<16 hold K 0..15;
// lanes>=16 hold K 16..31. Source: LDS row-major [16][Kpad] bf16.
__device__ __forceinline__ v16bf load_a_frag(const bf16* A, int Kpad, int kstep, int lane) {
  const int m = lane & 15, hi = lane >> 4;
  const bf16* p = A + m * Kpad + kstep * 32;
  v8bf c0 = *(const v8bf*)(p + hi * 16);
  v8bf c1 = *(const v8bf*)(p + hi * 16 + 8);
  v16bf r;
#pragma unroll
  for (int i = 0; i < 8; ++i) { r[i] = c0[i]; r[i + 8] = c1[i]; }
  return r;
}

// B-matrix 32x16 bf16: column n=l&15, K-half = (l>>4)*16; W' stored [N][Kpad]
// bf16 row-major so each lane reads 32 contiguous bytes (resident in L2/WGP$).
__device__ __forceinline__ v16bf load_b_frag(const bf16* W, int Kpad, int ntile, int kstep, int lane) {
  const int n = ntile * 16 + (lane & 15);
  const int k = kstep * 32 + (lane >> 4) * 16;
  return *(const v16bf*)(W + n * Kpad + k);
}

__device__ __forceinline__ v8f wmma_bf16(v16bf a, v16bf b, v8f c) {
  return __builtin_amdgcn_wmma_f32_16x16x32_bf16(false, a, false, b, (short)0, c, false, false);
}

// D layout: VGPR v -> M = v + 8*(lane>=16), N = lane&15. Store bias+ReLU+cvt.
__device__ __forceinline__ void store_d(v8f c, bf16* Dst, int Cpad, int dstcol,
                                        const float* bias_tile, int lane, bool do_relu) {
  const int n = lane & 15, hi = lane >> 4;
  const float bv = bias_tile[n];
#pragma unroll
  for (int v = 0; v < 8; ++v) {
    float val = c[v] + bv;
    if (do_relu) val = val > 0.f ? val : 0.f;
    Dst[(v + 8 * hi) * Cpad + dstcol + n] = (bf16)val;
  }
}

#if USE_TDM
// ---- Tensor Data Mover: DMA a 2D f32 tile (rows x cols, contiguous rows) ----
// D# per CDNA5 ISA ch.8: group0 = {flags, lds_addr, global_addr, type=2},
// group1 = {data_size, tensor dims/strides, tile dims}. 2D tile -> groups 2/3 zero.
__device__ __forceinline__ void tdm_load_tile_f32(const float* gsrc, void* lds_dst,
                                                  unsigned rows, unsigned cols,
                                                  unsigned tensor_rows) {
  const unsigned long long ga = (unsigned long long)(uintptr_t)gsrc;
  const unsigned lds_addr = (unsigned)(uintptr_t)lds_dst;  // low 32 bits = LDS offset

  u32x4 g0;
  g0[0] = 1u;                                        // count=1, no gather
  g0[1] = lds_addr;                                  // lds_addr [63:32]
  g0[2] = (unsigned)(ga & 0xFFFFFFFFu);              // global_addr low
  g0[3] = (unsigned)((ga >> 32) & 0x01FFFFFFu) | (2u << 30);  // addr hi | type=2

  const unsigned td0 = cols;           // tensor_dim0 (elements)
  const unsigned td1 = tensor_rows;    // tensor_dim1
  const unsigned long long st0 = cols; // tensor_dim0_stride

  i32x8 g1;
  g1[0] = (int)(2u << 16);                                   // data_size=2 (4 bytes)
  g1[1] = (int)((td0 & 0xFFFFu) << 16);                      // tensor_dim0[15:0]
  g1[2] = (int)((td0 >> 16) | ((td1 & 0xFFFFu) << 16));      // dim0 hi | dim1 lo
  g1[3] = (int)((td1 >> 16) | (cols << 16));                 // dim1 hi | tile_dim0
  g1[4] = (int)rows;                                         // tile_dim1 | tile_dim2=0
  g1[5] = (int)(st0 & 0xFFFFFFFFu);                          // dim0_stride lo
  g1[6] = (int)((st0 >> 32) & 0xFFFFu);                      // dim0_stride hi
  g1[7] = 0;

  const i32x4 z4 = {0, 0, 0, 0};
#if defined(__clang_major__) && (__clang_major__ >= 23)
  const i32x8 z8 = {0, 0, 0, 0, 0, 0, 0, 0};
  __builtin_amdgcn_tensor_load_to_lds(g0, g1, z4, z4, z8, 0);
#else
  __builtin_amdgcn_tensor_load_to_lds(g0, g1, z4, z4, 0);
#endif
}
#endif

// ---------------- weight prep: f32 [N][K] -> bf16 [N][Kpad] zero-padded ----------------
__device__ void conv_pad(const float* src, bf16* dst, int N, int K, int Kpad, int t, int stride) {
  const int total = N * Kpad;
  for (int i = t; i < total; i += stride) {
    const int n = i / Kpad, k = i % Kpad;
    dst[i] = (bf16)((k < K) ? src[n * K + k] : 0.0f);
  }
}

__global__ void prep_weights_kernel(PrepParams P, bf16* ws) {
  const int t = blockIdx.x * blockDim.x + threadIdx.x;
  const int stride = gridDim.x * blockDim.x;
#pragma unroll
  for (int p = 0; p < 2; ++p) {
    conv_pad(P.rw1[p], ws + WS_RW1(p), 64, 14, 32, t, stride);
    conv_pad(P.rw2[p], ws + WS_RW2(p), 32, 64, 64, t, stride);
    conv_pad(P.ow1[p], ws + WS_OW1(p), 64, 18, 32, t, stride);
    conv_pad(P.ow2[p], ws + WS_OW2(p), 32, 64, 64, t, stride);
  }
  conv_pad(P.qw1, ws + WS_QW1, 128, 128, 128, t, stride);
  conv_pad(P.qw2, ws + WS_QW2, 128, 128, 128, t, stride);
}

// ---------------- fused critic kernel ----------------
__global__ __launch_bounds__(WAVES * 32, 1)
void critic_crossattn_kernel(Params P, const bf16* __restrict__ ws) {
  __shared__ alignas(32) bf16 sOG[WAVES][96][32];      // object+goal feats, K padded
  __shared__ alignas(32) bf16 sOE[WAVES][96][32];      // object embeddings (per attn head)
  __shared__ alignas(32) bf16 sHID[WAVES][16][64];     // MLP hidden staging
  __shared__ alignas(32) bf16 sR[WAVES][2][16][32];    // robot feats r1/r2
  // Pool overlays: phase 1 uses it as the raw f32 x-tile (TDM destination);
  // phases 2-3 use it as the two bf16 trunk activation buffers. Disjoint
  // lifetimes separated by barriers.
  __shared__ alignas(64) unsigned char sPool[ROWS_PER_BLOCK * 128 * sizeof(float)];
  __shared__ float sDOT[WAVES][16][N_OBJ];
  __shared__ float sWGT[WAVES][16][N_OBJ];
  __shared__ float sRED[WAVES][16][2][2];

  float (*sXF)[128]    = reinterpret_cast<float(*)[128]>(sPool);               // [32][128] f32
  bf16 (*sH)[16][128]  = reinterpret_cast<bf16(*)[16][128]>(sPool);            // [WAVES][16][128]
  bf16 (*sH2)[16][128] = reinterpret_cast<bf16(*)[16][128]>(sPool + (size_t)WAVES * 16 * 128 * sizeof(bf16));

  const int lane = threadIdx.x & 31;
  const int w    = threadIdx.x >> 5;
  const int r    = lane & 15;
  const int hi   = lane >> 4;
  const int br   = w * 16 + r;                       // row within block tile
  const int gr   = blockIdx.x * ROWS_PER_BLOCK + br; // global batch row

  __builtin_prefetch(ws + ((threadIdx.x & 63) << 6), 0, 0);  // warm weight lines

  // ---------- Phase 1: stage inputs as bf16 A-tiles in LDS ----------
  for (int i = lane; i < 96 * 32; i += 32) ((bf16*)sOG[w])[i] = (bf16)0.0f;
  for (int i = lane; i < 2 * 16 * 32; i += 32) ((bf16*)sR[w])[i] = (bf16)0.0f;

#if USE_TDM
  // One TDM op DMAs the whole contiguous 32x128 f32 x-tile into LDS.
  if (threadIdx.x == 0) {
    tdm_load_tile_f32(P.x + (size_t)blockIdx.x * ROWS_PER_BLOCK * 128, sPool,
                      ROWS_PER_BLOCK, 128, BATCH);
    __builtin_amdgcn_s_wait_tensorcnt(0);
  }
  __syncthreads();
  const float* xr = sXF[br];                   // LDS-resident f32 row
#else
  const float* xr = P.x + (size_t)gr * 128;    // fallback: direct global row
#endif
  const float* ar = P.act + (size_t)gr * 8;

  if (hi == 0) {
    for (int i = 0; i < 10; ++i) sR[w][0][r][i] = (bf16)xr[i];
    for (int i = 0; i < 4; ++i)  sR[w][0][r][10 + i] = (bf16)ar[i];
  } else {
    for (int i = 0; i < 10; ++i) sR[w][1][r][i] = (bf16)xr[10 + i];
    for (int i = 0; i < 4; ++i)  sR[w][1][r][10 + i] = (bf16)ar[4 + i];
  }
  for (int o = hi * 3; o < hi * 3 + 3; ++o) {
    bf16* dst = &sOG[w][r * N_OBJ + o][0];
    for (int i = 0; i < 15; ++i) dst[i] = (bf16)xr[20 + 15 * o + i];
    for (int i = 0; i < 3; ++i)  dst[15 + i] = (bf16)xr[110 + 3 * o + i];
  }
  __syncthreads();   // x-tile fully consumed; pool may be reused as sH/sH2

  // ---------- Phase 2: two attention heads ----------
  for (int p = 0; p < 2; ++p) {
    // robot MLP: H1 = relu(r @ rw1^T + rb1)  [16x64]
    {
      v16bf a = load_a_frag(&sR[w][p][0][0], 32, 0, lane);
#pragma unroll
      for (int nt = 0; nt < 4; ++nt) {
        v16bf b = load_b_frag(ws + WS_RW1(p), 32, nt, 0, lane);
        v8f c = {};
        c = wmma_bf16(a, b, c);
        store_d(c, &sHID[w][0][0], 64, nt * 16, P.rb1[p] + nt * 16, lane, true);
      }
    }
    __syncthreads();
    // RE = relu(H1 @ rw2^T + rb2)  [16x32] -> sH cols p*64..+31
#pragma unroll
    for (int nt = 0; nt < 2; ++nt) {
      v8f c = {};
#pragma unroll
      for (int ks = 0; ks < 2; ++ks) {
        v16bf a = load_a_frag(&sHID[w][0][0], 64, ks, lane);
        v16bf b = load_b_frag(ws + WS_RW2(p), 64, nt, ks, lane);
        c = wmma_bf16(a, b, c);
      }
      store_d(c, &sH[w][0][0], 128, p * 64 + nt * 16, P.rb2[p] + nt * 16, lane, true);
    }
    __syncthreads();

    // object MLP over 6 tiles of 16 (row,obj) pairs
    for (int mt = 0; mt < 6; ++mt) {
      v16bf a = load_a_frag(&sOG[w][mt * 16][0], 32, 0, lane);
#pragma unroll
      for (int nt = 0; nt < 4; ++nt) {
        v16bf b = load_b_frag(ws + WS_OW1(p), 32, nt, 0, lane);
        v8f c = {};
        c = wmma_bf16(a, b, c);
        store_d(c, &sHID[w][0][0], 64, nt * 16, P.ob1[p] + nt * 16, lane, true);
      }
      __syncthreads();
#pragma unroll
      for (int nt = 0; nt < 2; ++nt) {
        v8f c = {};
#pragma unroll
        for (int ks = 0; ks < 2; ++ks) {
          v16bf aa = load_a_frag(&sHID[w][0][0], 64, ks, lane);
          v16bf b  = load_b_frag(ws + WS_OW2(p), 64, nt, ks, lane);
          c = wmma_bf16(aa, b, c);
        }
        store_d(c, &sOE[w][mt * 16][0], 32, nt * 16, P.ob2[p] + nt * 16, lane, true);
      }
      __syncthreads();
    }

    // attention: dots, softmax over 6 objects, weighted sum, LayerNorm, ReLU
    for (int o = hi * 3; o < hi * 3 + 3; ++o) {
      float d = 0.f;
      for (int k = 0; k < 32; ++k)
        d += (float)sH[w][r][p * 64 + k] * (float)sOE[w][r * N_OBJ + o][k];
      sDOT[w][r][o] = d * 0.17677669529663689f;  // 1/sqrt(32)
    }
    __syncthreads();
    if (hi == 0) {
      float mx = -3.4e38f;
      for (int o = 0; o < N_OBJ; ++o) mx = fmaxf(mx, sDOT[w][r][o]);
      float s = 0.f;
      float e[N_OBJ];
      for (int o = 0; o < N_OBJ; ++o) { e[o] = __expf(sDOT[w][r][o] - mx); s += e[o]; }
      const float inv = 1.0f / s;
      for (int o = 0; o < N_OBJ; ++o) sWGT[w][r][o] = e[o] * inv;
    }
    __syncthreads();
    float wf[16];
    float s1 = 0.f, s2 = 0.f;
    for (int j = 0; j < 16; ++j) {
      const int d = hi * 16 + j;
      float v = 0.f;
      for (int o = 0; o < N_OBJ; ++o)
        v += sWGT[w][r][o] * (float)sOE[w][r * N_OBJ + o][d];
      wf[j] = v; s1 += v; s2 += v * v;
    }
    sRED[w][r][hi][0] = s1; sRED[w][r][hi][1] = s2;
    __syncthreads();
    const float sum = sRED[w][r][0][0] + sRED[w][r][1][0];
    const float sq  = sRED[w][r][0][1] + sRED[w][r][1][1];
    const float mu  = sum * (1.0f / 32.0f);
    const float var = sq * (1.0f / 32.0f) - mu * mu;
    const float inv = rsqrtf(var + 1e-5f);
    for (int j = 0; j < 16; ++j) {
      const int d = hi * 16 + j;
      float val = (wf[j] - mu) * inv * P.g[p][d] + P.b[p][d];
      val = val > 0.f ? val : 0.f;
      sH[w][r][p * 64 + 32 + d] = (bf16)val;
    }
    __syncthreads();
  }

  // ---------- Phase 3: trunk MLP 128 -> 128 -> 128 -> 1 ----------
  {
    v16bf a[4];
#pragma unroll
    for (int ks = 0; ks < 4; ++ks) a[ks] = load_a_frag(&sH[w][0][0], 128, ks, lane);
#pragma unroll
    for (int nt = 0; nt < 8; ++nt) {
      v8f c = {};
#pragma unroll
      for (int ks = 0; ks < 4; ++ks) {
        v16bf b = load_b_frag(ws + WS_QW1, 128, nt, ks, lane);
        c = wmma_bf16(a[ks], b, c);
      }
      store_d(c, &sH2[w][0][0], 128, nt * 16, P.qb1 + nt * 16, lane, true);
    }
  }
  __syncthreads();
  {
    v16bf a[4];
#pragma unroll
    for (int ks = 0; ks < 4; ++ks) a[ks] = load_a_frag(&sH2[w][0][0], 128, ks, lane);
#pragma unroll
    for (int nt = 0; nt < 8; ++nt) {
      v8f c = {};
#pragma unroll
      for (int ks = 0; ks < 4; ++ks) {
        v16bf b = load_b_frag(ws + WS_QW2, 128, nt, ks, lane);
        c = wmma_bf16(a[ks], b, c);
      }
      store_d(c, &sH[w][0][0], 128, nt * 16, P.qb2 + nt * 16, lane, true);
    }
  }
  __syncthreads();
  // final 128 -> 1 dot (N=1: VALU cheaper than a padded WMMA tile)
  {
    float acc = 0.f;
    for (int k = 0; k < 64; ++k)
      acc += (float)sH[w][r][hi * 64 + k] * P.qw3[hi * 64 + k];
    sRED[w][r][hi][0] = acc;
    __syncthreads();
    if (hi == 0)
      P.out[gr] = sRED[w][r][0][0] + sRED[w][r][1][0] + P.qb3[0];
  }
}

extern "C" void kernel_launch(void* const* d_in, const int* in_sizes, int n_in,
                              void* d_out, int out_size, void* d_ws, size_t ws_size,
                              hipStream_t stream) {
  (void)in_sizes; (void)n_in; (void)out_size; (void)ws_size;
  auto f = [&](int i) { return (const float*)d_in[i]; };

  bf16* ws = (bf16*)d_ws;

  PrepParams pp;
  pp.rw1[0] = f(2);  pp.rw2[0] = f(4);  pp.ow1[0] = f(6);  pp.ow2[0] = f(8);
  pp.rw1[1] = f(12); pp.rw2[1] = f(14); pp.ow1[1] = f(16); pp.ow2[1] = f(18);
  pp.qw1 = f(22);    pp.qw2 = f(24);
  prep_weights_kernel<<<64, 256, 0, stream>>>(pp, ws);

  Params P;
  P.x = f(0); P.act = f(1);
  P.rb1[0] = f(3);  P.rb2[0] = f(5);  P.ob1[0] = f(7);  P.ob2[0] = f(9);
  P.g[0] = f(10);   P.b[0] = f(11);
  P.rb1[1] = f(13); P.rb2[1] = f(15); P.ob1[1] = f(17); P.ob2[1] = f(19);
  P.qb1 = f(23); P.qb2 = f(25); P.qw3 = f(26); P.qb3 = f(27);
  P.out = (float*)d_out;

  critic_crossattn_kernel<<<BATCH / ROWS_PER_BLOCK, WAVES * 32, 0, stream>>>(P, ws);
}
